// TransducerJoint_22960895165074
// MI455X (gfx1250) — matrix operand order repaired and checked by reading, compiled, not run on hardware
//
#include <hip/hip_runtime.h>

typedef __attribute__((ext_vector_type(16))) _Float16 v16h;
typedef __attribute__((ext_vector_type(8)))  float    v8f;
typedef __attribute__((ext_vector_type(4)))  unsigned int u32x4;
typedef __attribute__((ext_vector_type(8)))  int          i32x8;
typedef __attribute__((ext_vector_type(4)))  int          i32x4;

#define Bk   4
#define Tk   300
#define Uk   100
#define Dk   512     // K dim (D_MODEL = D_E = D_P)
#define Vk   1024    // vocab
#define VCH  128     // vocab columns per joint block
#define HPAD 8       // half-element pad (16B) to break LDS bank conflicts

#if defined(__has_builtin)
#if __has_builtin(__builtin_amdgcn_tensor_load_to_lds) && __has_builtin(__builtin_amdgcn_s_wait_tensorcnt)
#define TJ_HAVE_TDM 1
#endif
#endif

union AFrag { v16h h; unsigned int u[8]; };

// Single-instruction CDNA5 transcendental tanh. The trailing s_delay_alu
// enforces the TRANS32 result-use hazard the compiler cannot see through asm.
__device__ __forceinline__ float tj_fast_tanh(float x) {
    float y;
    asm("v_tanh_f32 %0, %1\n\ts_delay_alu instid0(TRANS32_DEP_1)"
        : "=v"(y) : "v"(x));
    return y;
}

// ---------------------------------------------------------------- conversions

__global__ void tj_f32_to_f16(const float* __restrict__ in, _Float16* __restrict__ out, int n) {
    int i = blockIdx.x * 256 + threadIdx.x;
    if (i < n) out[i] = (_Float16)in[i];
}

// in: [K][N] f32 row-major  ->  out: [N][K] f16 row-major (transposed)
__global__ void tj_transpose_to_f16(const float* __restrict__ in, _Float16* __restrict__ out,
                                    int K, int N) {
    int i = blockIdx.x * 256 + threadIdx.x;
    if (i < K * N) {
        int k = i / N, n = i % N;
        out[n * K + k] = (_Float16)in[i];
    }
}

// ---------------------------------------------------------------- projection
// Y[M][512] = Xh[M][512] (f16) @ Wt^T  + bias   (Wt stored [N=512][K=512] f16)
#define VCHP 128

__global__ __launch_bounds__(256) void tj_proj_wmma(
    const _Float16* __restrict__ Xh,
    const _Float16* __restrict__ Wt,
    const float*    __restrict__ bias,
    float*          __restrict__ Y) {

    __shared__ _Float16 sA[16][Dk + HPAD];

    const int tid = threadIdx.x;
    const int m0  = blockIdx.x * 16;
    const int n0  = blockIdx.y * VCHP;

    for (int i = tid; i < 16 * Dk / 8; i += 256) {
        int row = (i * 8) >> 9;
        int col = (i * 8) & (Dk - 1);
        *(uint4*)&sA[row][col] = *(const uint4*)&Xh[(size_t)(m0 + row) * Dk + col];
    }
    __syncthreads();

    const int wave = tid >> 5;
    const int lane = tid & 31;
    const int nloc = lane & 15;
    const int hi   = lane >> 4;

    const int ncol = n0 + wave * 16 + nloc;
    const _Float16* brow = &Wt[(size_t)ncol * Dk];
    const _Float16* arow = &sA[nloc][0];

    v8f acc = {};
    for (int kc = 0; kc < Dk; kc += 32) {
        AFrag a, b;
#pragma unroll
        for (int r = 0; r < 8; ++r) {
            int ka = kc + ((r & 4) << 2) + hi * 8 + ((r & 3) << 1); // A 16x32 f16 layout
            a.u[r] = *(const unsigned int*)(arow + ka);
            int kb = kc + ((r & 4) << 2) + hi * 8 + ((r & 3) << 1); // B 32x16 f16 layout
            b.u[r] = *(const unsigned int*)(brow + kb);
        }
        acc = __builtin_amdgcn_wmma_f32_16x16x32_f16(false, a.h, false, b.h,
                                                     (short)0, acc, false, false);
    }

    const float bv = bias[ncol];
#pragma unroll
    for (int r = 0; r < 8; ++r) {
        int m = hi * 8 + r;                 // C/D layout: M = r (+8 for hi lanes)
        Y[(size_t)(m0 + m) * Dk + ncol] = acc[r] + bv;
    }
}

// ---------------------------------------------------------------- joint
// out[b,t,u,v] = tanh(F[b,t,:] + G[b,u,:]) @ Wjt^T + b_joint
// LDS = 130KB W chunk + 16.3KB tanh tile < 160KB  =>  2 blocks per WGP so
// one block's tanh (VALU/TRANS) overlaps the other's WMMA (XDL).

__global__ __launch_bounds__(256) void tj_joint_wmma(
    const float*    __restrict__ F,      // [B*T][512] f32
    const float*    __restrict__ G,      // [B*U][512] f32
    const _Float16* __restrict__ Wjt,    // [1024][512] f16 (transposed)
    const float*    __restrict__ bj,     // [1024]
    float*          __restrict__ out) {  // [B,T,U,1024] f32

    __shared__ _Float16 sW[VCH][Dk + HPAD];   // 128 x 520 halfs = 130 KB
    __shared__ _Float16 sH[16][Dk + HPAD];    // tanh tile, f16

    const int tid = threadIdx.x;
    const int v0  = blockIdx.x * VCH;
    const int u0  = blockIdx.y * 16;
    const int b   = blockIdx.z;

    const int wave = tid >> 5;
    const int lane = tid & 31;

    // ---- stage 128x512 W_joint chunk into LDS once (reused for all 300 t)
#if TJ_HAVE_TDM
    if (tid < 32) {   // wave 0 issues one TDM op; EXEC is ignored by TDM
        unsigned long long ga = (unsigned long long)(size_t)&Wjt[(size_t)v0 * Dk];
        unsigned ldsoff = (unsigned)(size_t)&sW[0][0];   // generic LDS ptr low bits = LDS offset
        u32x4 g0;
        g0[0] = 1u;                                      // count=1, user-mode load
        g0[1] = ldsoff;                                  // lds_addr
        g0[2] = (unsigned)(ga & 0xffffffffu);            // global_addr[31:0]
        g0[3] = (unsigned)((ga >> 32) & 0x01ffffffu)     // global_addr[56:32]
              | (2u << 30);                              // type = 2 ("image")
        i32x8 g1;
        g1[0] = (int)((1u << 16)        // data_size = 2 bytes
                    | (1u << 20)        // pad_enable
                    | (7u << 22)        // pad_interval = 256 dwords (one 1024B row)
                    | (3u << 25));      // pad_amount = 4 dwords (16B = HPAD)
        g1[1] = (int)(512u << 16);      // tensor_dim0[15:0]=512 at bits[63:48]
        g1[2] = (int)(128u << 16);      // tensor_dim0 hi=0 | tensor_dim1[15:0]=128
        g1[3] = (int)(512u << 16);      // tensor_dim1 hi=0 | tile_dim0=512
        g1[4] = (int)(128u);            // tile_dim1=128, tile_dim2=0 (2D)
        g1[5] = (int)(512u);            // tensor_dim0_stride[31:0]=512 elems
        g1[6] = 0;
        g1[7] = 0;
        i32x4 gz = {0, 0, 0, 0};
#if __clang_major__ >= 23
        i32x8 gz8 = {0, 0, 0, 0, 0, 0, 0, 0};
        __builtin_amdgcn_tensor_load_to_lds(g0, g1, gz, gz, gz8, 0);
#else
        __builtin_amdgcn_tensor_load_to_lds(g0, g1, gz, gz, 0);
#endif
        __builtin_amdgcn_s_wait_tensorcnt(0);
    }
#else
    for (int i = tid; i < VCH * Dk / 8; i += 256) {
        int row = (i * 8) >> 9;
        int col = (i * 8) & (Dk - 1);
        *(uint4*)&sW[row][col] = *(const uint4*)&Wjt[(size_t)(v0 + row) * Dk + col];
    }
#endif

    // ---- load this thread's 2 rows x 16 cols of g into registers (no LDS)
    // wave w owns tanh rows {2w, 2w+1}; lane l owns cols jj*128 + l*4 (jj=0..3)
    float gr[2][16];
#pragma unroll
    for (int rr = 0; rr < 2; ++rr) {
        int u = u0 + wave * 2 + rr;
#pragma unroll
        for (int jj = 0; jj < 4; ++jj) {
            float4 gv = make_float4(0.f, 0.f, 0.f, 0.f);
            if (u < Uk)
                gv = *(const float4*)&G[(size_t)(b * Uk + u) * Dk + jj * 128 + lane * 4];
            gr[rr][jj * 4 + 0] = gv.x;
            gr[rr][jj * 4 + 1] = gv.y;
            gr[rr][jj * 4 + 2] = gv.z;
            gr[rr][jj * 4 + 3] = gv.w;
        }
    }
    __syncthreads();

    const int nloc = lane & 15;
    const int hi   = lane >> 4;
    const int ncol = v0 + wave * 16 + nloc;
    const float bias = bj[ncol];
    const _Float16* brow = &sW[wave * 16 + nloc][0];
    const _Float16* arow = &sH[nloc][0];
    const bool full = (u0 + 16 <= Uk);

    for (int t = 0; t < Tk; ++t) {
        const float* frow = F + (size_t)(b * Tk + t) * Dk;
        if (t + 1 < Tk) __builtin_prefetch(frow + Dk + (tid & 63) * 8, 0, 1);

        // compute sH = (f16) tanh(f + g) using v_tanh_f32; fully coalesced /
        // bank-conflict-free mapping (lane-contiguous 16B f loads, 8B ds stores)
#pragma unroll
        for (int rr = 0; rr < 2; ++rr) {
            int row = wave * 2 + rr;
#pragma unroll
            for (int jj = 0; jj < 4; ++jj) {
                int c = jj * 128 + lane * 4;
                float4 fv = *(const float4*)&frow[c];
                union { _Float16 h[4]; unsigned long long q; } pk;
                pk.h[0] = (_Float16)tj_fast_tanh(fv.x + gr[rr][jj * 4 + 0]);
                pk.h[1] = (_Float16)tj_fast_tanh(fv.y + gr[rr][jj * 4 + 1]);
                pk.h[2] = (_Float16)tj_fast_tanh(fv.z + gr[rr][jj * 4 + 2]);
                pk.h[3] = (_Float16)tj_fast_tanh(fv.w + gr[rr][jj * 4 + 3]);
                *(unsigned long long*)&sH[row][c] = pk.q;
            }
        }
        __syncthreads();

        v8f acc = {};
        for (int kc = 0; kc < Dk; kc += 32) {
            AFrag a, bb;
#pragma unroll
            for (int r = 0; r < 8; ++r) {
                int ka = kc + ((r & 4) << 2) + hi * 8 + ((r & 3) << 1);
                a.u[r] = *(const unsigned int*)(arow + ka);
                int kb = kc + ((r & 4) << 2) + hi * 8 + ((r & 3) << 1);
                bb.u[r] = *(const unsigned int*)(brow + kb);
            }
            acc = __builtin_amdgcn_wmma_f32_16x16x32_f16(false, a.h, false, bb.h,
                                                         (short)0, acc, false, false);
        }

        const size_t rowbase = (size_t)(b * Tk + t) * Uk;
        if (full) {
#pragma unroll
            for (int r = 0; r < 8; ++r)
                out[(rowbase + u0 + hi * 8 + r) * Vk + ncol] = acc[r] + bias;
        } else {
#pragma unroll
            for (int r = 0; r < 8; ++r) {
                int u = u0 + hi * 8 + r;
                if (u < Uk) out[(rowbase + u) * Vk + ncol] = acc[r] + bias;
            }
        }
        __syncthreads();   // protect sH before next t overwrites it
    }
}

// ---------------------------------------------------------------- launch

extern "C" void kernel_launch(void* const* d_in, const int* in_sizes, int n_in,
                              void* d_out, int out_size, void* d_ws, size_t ws_size,
                              hipStream_t stream) {
    const float* enc     = (const float*)d_in[0];   // [4,300,512]
    const float* pred    = (const float*)d_in[1];   // [4,100,512]
    const float* W_enc   = (const float*)d_in[2];   // [512,512]
    const float* b_enc   = (const float*)d_in[3];   // [512]
    const float* W_pred  = (const float*)d_in[4];   // [512,512]
    const float* b_pred  = (const float*)d_in[5];   // [512]
    const float* W_joint = (const float*)d_in[6];   // [512,1024]
    const float* b_joint = (const float*)d_in[7];   // [1024]
    float* out = (float*)d_out;

    char* ws = (char*)d_ws;
    size_t off = 0;
    auto take = [&](size_t bytes) -> void* {
        off = (off + 255) & ~(size_t)255;
        void* p = ws + off;
        off += bytes;
        return p;
    };

    const int Menc = Bk * Tk;   // 1200
    const int Mprd = Bk * Uk;   // 400

    _Float16* enc_h   = (_Float16*)take((size_t)Menc * Dk * 2);
    _Float16* pred_h  = (_Float16*)take((size_t)Mprd * Dk * 2);
    _Float16* Wenc_t  = (_Float16*)take((size_t)Dk * Dk * 2);
    _Float16* Wpred_t = (_Float16*)take((size_t)Dk * Dk * 2);
    _Float16* Wjnt_t  = (_Float16*)take((size_t)Vk * Dk * 2);
    float*    Fbuf    = (float*)take((size_t)Menc * Dk * 4);
    float*    Gbuf    = (float*)take((size_t)Mprd * Dk * 4);

    tj_f32_to_f16<<<(Menc * Dk + 255) / 256, 256, 0, stream>>>(enc, enc_h, Menc * Dk);
    tj_f32_to_f16<<<(Mprd * Dk + 255) / 256, 256, 0, stream>>>(pred, pred_h, Mprd * Dk);
    tj_transpose_to_f16<<<(Dk * Dk + 255) / 256, 256, 0, stream>>>(W_enc, Wenc_t, Dk, Dk);
    tj_transpose_to_f16<<<(Dk * Dk + 255) / 256, 256, 0, stream>>>(W_pred, Wpred_t, Dk, Dk);
    tj_transpose_to_f16<<<(Dk * Vk + 255) / 256, 256, 0, stream>>>(W_joint, Wjnt_t, Dk, Vk);

    tj_proj_wmma<<<dim3(Menc / 16, Dk / VCHP), 256, 0, stream>>>(enc_h, Wenc_t, b_enc, Fbuf);
    tj_proj_wmma<<<dim3(Mprd / 16, Dk / VCHP), 256, 0, stream>>>(pred_h, Wpred_t, b_pred, Gbuf);

    tj_joint_wmma<<<dim3(Vk / VCH, (Uk + 15) / 16, Bk), 256, 0, stream>>>(
        Fbuf, Gbuf, Wjnt_t, b_joint, out);
}
